// EdgeGAT_GRU_8650064134835
// MI455X (gfx1250) — matrix operand order repaired and checked by reading, not compiled, not run on hardware
//
#include <hip/hip_runtime.h>
#include <hip/hip_bf16.h>
#include <math.h>

// Problem constants (from reference setup_inputs)
#define N_NODES 20000
#define N_EDGES 65536
#define HDIM    256     // H
#define GDIM    256     // G
#define K0PAD   544     // 2H+3 = 515 padded up to a multiple of 32
#define TGATE   768     // 3*G
#define BN_EPS  1e-5f
#define LRELU_S 0.2f

typedef __bf16 bf16_t;
typedef bf16_t v8bf  __attribute__((ext_vector_type(8)));
typedef bf16_t v16bf __attribute__((ext_vector_type(16)));
typedef float  v8f   __attribute__((ext_vector_type(8)));

// ---------- helpers ----------
__device__ __forceinline__ unsigned short f32_to_bf16_rte(float f) {
  unsigned u = __float_as_uint(f);
  u += 0x7FFFu + ((u >> 16) & 1u);          // round-to-nearest-even
  return (unsigned short)(u >> 16);
}
// monotone float<->uint mapping for atomicMax-based segment max
__device__ __forceinline__ unsigned map_f32(float f) {
  unsigned u = __float_as_uint(f);
  return (u & 0x80000000u) ? ~u : (u | 0x80000000u);
}
__device__ __forceinline__ float unmap_f32(unsigned m) {
  unsigned u = (m & 0x80000000u) ? (m & 0x7FFFFFFFu) : ~m;
  return __uint_as_float(u);
}
__device__ __forceinline__ float sigmoidf_(float x) { return 1.0f / (1.0f + __expf(-x)); }

__device__ __forceinline__ v16bf cat16(v8bf lo, v8bf hi) {
  return __builtin_shufflevector(lo, hi, 0,1,2,3,4,5,6,7,8,9,10,11,12,13,14,15);
}

// ---------- generic utility kernels ----------
__global__ void zero_u32(unsigned* __restrict__ p, unsigned n) {
  unsigned i = blockIdx.x * blockDim.x + threadIdx.x;
  if (i < n) p[i] = 0u;
}

// elementwise f32 -> bf16 (for W2 which is already [K,N] row-major)
__global__ void conv_bf16(const float* __restrict__ src, unsigned short* __restrict__ dst, unsigned n) {
  unsigned i = blockIdx.x * blockDim.x + threadIdx.x;
  if (i < n) dst[i] = f32_to_bf16_rte(src[i]);
}

// W [nout, kin] f32 (torch Linear layout) -> B [Kpad, nout] bf16, zero-padded rows
__global__ void transpose_w(const float* __restrict__ W, unsigned short* __restrict__ B,
                            int nout, int kin, int kpad) {
  int idx = blockIdx.x * blockDim.x + threadIdx.x;
  if (idx >= kpad * nout) return;
  int k = idx / nout, n = idx - k * nout;
  B[idx] = (k < kin) ? f32_to_bf16_rte(W[(size_t)n * kin + k]) : (unsigned short)0;
}

// ---------- WMMA bf16 GEMM: C[M,N] = A[M,K] @ B[K,N] (+bias, opt relu) ----------
// Register-blocked: each wave computes a 32x64 block (2 M-tiles x 4 N-tiles,
// 8 f32 accumulators), so one K-step issues 4 A-frag b128 loads + 64 strided 16-bit B loads for
// 8 v_wmma_f32_16x16x32_bf16. Block = 2 waves along N.
// grid = (Nout/128, M/32). Requires M%32==0, Nout%128==0, K%32==0 (all hold).
#define TM 2
#define TN 4
__global__ void wmma_gemm_bf16(const unsigned short* __restrict__ A, int lda,
                               const unsigned short* __restrict__ B, int ldb,
                               const float* __restrict__ bias,
                               float* __restrict__ C, int ldc,
                               int K, int relu) {
  const int lane  = threadIdx.x & 31;
  const int wv    = threadIdx.x >> 5;                 // 0..1
  const int mBase = blockIdx.y * (TM * 16);
  const int nBase = (blockIdx.x * 2 + wv) * (TN * 16);
  // A fragment mapping (ISA 7.12.2, 16-bit A 16x32): lanes 0-15 row M=lane, K 0..7 / 16..23;
  // lanes 16-31 same rows, K 8..15 / 24..31.
  const int rowIn = lane & 15;
  const int koff  = (lane < 16) ? 0 : 8;
  const unsigned short* Ap0 = A + (size_t)(mBase + rowIn) * lda + koff;
  const unsigned short* Ap1 = A + (size_t)(mBase + 16 + rowIn) * lda + koff;
  // B fragment (dense 32x16 16-bit): lane L holds column N=L%16, K 0..7 / 16..23 (lanes 0-15), K 8..15 / 24..31 (lanes 16-31)
  const unsigned short* Bp  = B + (size_t)((lane < 16) ? 0 : 8) * ldb + nBase + (lane & 15);
  const size_t bStep = (size_t)32 * ldb;

  v8f acc[TM][TN];
  #pragma unroll
  for (int i = 0; i < TM; ++i)
    #pragma unroll
    for (int j = 0; j < TN; ++j)
      acc[i][j] = (v8f){0.f, 0.f, 0.f, 0.f, 0.f, 0.f, 0.f, 0.f};

  for (int kk = K >> 5; kk > 0; --kk) {
    v16bf a0 = cat16(*(const v8bf*)(Ap0), *(const v8bf*)(Ap0 + 16));
    v16bf a1 = cat16(*(const v8bf*)(Ap1), *(const v8bf*)(Ap1 + 16));
    Ap0 += 32; Ap1 += 32;
    #pragma unroll
    for (int j = 0; j < TN; ++j) {
      v16bf bf;
      #pragma unroll
      for (int s = 0; s < 16; ++s)
        bf[s] = __builtin_bit_cast(bf16_t, Bp[(size_t)((s < 8) ? s : s + 8) * ldb + j * 16]);
      acc[0][j] = __builtin_amdgcn_wmma_f32_16x16x32_bf16(false, a0, false, bf, (short)0,
                                                          acc[0][j], false, false);
      acc[1][j] = __builtin_amdgcn_wmma_f32_16x16x32_bf16(false, a1, false, bf, (short)0,
                                                          acc[1][j], false, false);
    }
    Bp += bStep;
  }

  // C/D layout: lane%16 = N col; VGPR r -> M = r (+8 for upper half-wave)
  const int nCol0 = nBase + (lane & 15);
  const int mOff0 = mBase + ((lane < 16) ? 0 : 8);
  #pragma unroll
  for (int j = 0; j < TN; ++j) {
    const int nCol = nCol0 + j * 16;
    const float bv = bias ? bias[nCol] : 0.f;
    #pragma unroll
    for (int i = 0; i < TM; ++i) {
      const int mOff = mOff0 + i * 16;
      #pragma unroll
      for (int r = 0; r < 8; ++r) {
        float v = acc[i][j][r] + bv;
        if (relu) v = fmaxf(v, 0.f);
        C[(size_t)(mOff + r) * ldc + nCol] = v;
      }
    }
  }
}

// ---------- GAT layer 1 (rank-1 fast path) ----------
__global__ void dot2_256(const float* __restrict__ w, const float* __restrict__ a1,
                         const float* __restrict__ a2, float* __restrict__ cbuf) {
  __shared__ float s1[256], s2[256];
  int t = threadIdx.x;
  s1[t] = w[t] * a1[t];
  s2[t] = w[t] * a2[t];
  __syncthreads();
  for (int s = 128; s > 0; s >>= 1) {
    if (t < s) { s1[t] += s1[t + s]; s2[t] += s2[t + s]; }
    __syncthreads();
  }
  if (t == 0) { cbuf[0] = s1[0]; cbuf[1] = s2[0]; }
}

__global__ void scale_scores(const float* __restrict__ x, const float* __restrict__ cbuf,
                             float* __restrict__ a_src, float* __restrict__ a_dst, int n) {
  int i = blockIdx.x * blockDim.x + threadIdx.x;
  if (i >= n) return;
  float xv = x[i];
  a_src[i] = xv * cbuf[0];
  a_dst[i] = xv * cbuf[1];
}

// ---------- edge softmax (shared by both GAT layers) ----------
__global__ void edge_lrelu_max(const float* __restrict__ as, const float* __restrict__ ad,
                               const int* __restrict__ src, const int* __restrict__ dst,
                               unsigned* __restrict__ mmap, int E, int N) {
  int i = blockIdx.x * blockDim.x + threadIdx.x;
  if (i >= E + N) return;
  int s = (i < E) ? src[i] : (i - E);
  int d = (i < E) ? dst[i] : (i - E);
  float e = as[s] + ad[d];
  e = (e > 0.f) ? e : LRELU_S * e;
  atomicMax(&mmap[d], map_f32(e));
}

__global__ void edge_exp_sum(const float* __restrict__ as, const float* __restrict__ ad,
                             const int* __restrict__ src, const int* __restrict__ dst,
                             const unsigned* __restrict__ mmap, float* __restrict__ ebuf,
                             float* __restrict__ z, int E, int N) {
  int i = blockIdx.x * blockDim.x + threadIdx.x;
  if (i >= E + N) return;
  int s = (i < E) ? src[i] : (i - E);
  int d = (i < E) ? dst[i] : (i - E);
  float e = as[s] + ad[d];
  e = (e > 0.f) ? e : LRELU_S * e;
  float ev = __expf(e - unmap_f32(mmap[d]));
  ebuf[i] = ev;
  atomicAdd(&z[d], ev);
}

__global__ void gat1_aggregate(const float* __restrict__ x, const int* __restrict__ src,
                               const int* __restrict__ dst, const float* __restrict__ ebuf,
                               const float* __restrict__ z, float* __restrict__ agg, int E, int N) {
  int i = blockIdx.x * blockDim.x + threadIdx.x;
  if (i >= E + N) return;
  int s = (i < E) ? src[i] : (i - E);
  int d = (i < E) ? dst[i] : (i - E);
  float alpha = ebuf[i] / z[d];
  atomicAdd(&agg[d], alpha * x[s]);
}

// h1_bf16[n,c] = bf16(relu(bn(agg[n]*W1[c] + b1[c])))
__global__ void gat1_finalize(const float* __restrict__ agg, const float* __restrict__ W1,
                              const float* __restrict__ b1, const float* __restrict__ gm,
                              const float* __restrict__ bt, const float* __restrict__ mu,
                              const float* __restrict__ vr, unsigned short* __restrict__ h1b) {
  int n = blockIdx.x, c = threadIdx.x;
  float v = agg[n] * W1[c] + b1[c];
  v = (v - mu[c]) * rsqrtf(vr[c] + BN_EPS) * gm[c] + bt[c];
  v = fmaxf(v, 0.f);
  h1b[(size_t)n * HDIM + c] = f32_to_bf16_rte(v);
}

// ---------- GAT layer 2 ----------
// per-node attention scores: o1[n]=dot(X[n,:],w1), o2[n]=dot(X[n,:],w2)
__global__ void rowdot2(const float* __restrict__ X, const float* __restrict__ w1,
                        const float* __restrict__ w2, float* __restrict__ o1,
                        float* __restrict__ o2) {
  __shared__ float s1[256], s2[256];
  int n = blockIdx.x, t = threadIdx.x;
  float v = X[(size_t)n * HDIM + t];
  s1[t] = v * w1[t];
  s2[t] = v * w2[t];
  __syncthreads();
  for (int s = 128; s > 0; s >>= 1) {
    if (t < s) { s1[t] += s1[t + s]; s2[t] += s2[t + s]; }
    __syncthreads();
  }
  if (t == 0) { o1[n] = s1[0]; o2[n] = s2[0]; }
}

__global__ void gat2_scatter(const float* __restrict__ xw2, const int* __restrict__ src,
                             const int* __restrict__ dst, const float* __restrict__ ebuf,
                             const float* __restrict__ z, float* __restrict__ acc, int E) {
  int i = blockIdx.x, c = threadIdx.x;
  int s = (i < E) ? src[i] : (i - E);
  int d = (i < E) ? dst[i] : (i - E);
  float alpha = ebuf[i] / z[d];
  atomicAdd(&acc[(size_t)d * HDIM + c], alpha * xw2[(size_t)s * HDIM + c]);
}

__global__ void gat2_finalize(const float* __restrict__ acc, const float* __restrict__ b2,
                              const float* __restrict__ gm, const float* __restrict__ bt,
                              const float* __restrict__ mu, const float* __restrict__ vr,
                              unsigned short* __restrict__ h2b) {
  int n = blockIdx.x, c = threadIdx.x;
  float v = acc[(size_t)n * HDIM + c] + b2[c];
  v = (v - mu[c]) * rsqrtf(vr[c] + BN_EPS) * gm[c] + bt[c];
  v = fmaxf(v, 0.f);
  h2b[(size_t)n * HDIM + c] = f32_to_bf16_rte(v);
}

// edge features: [h2[src] | h2[dst] | edge_attr | 0-pad] in bf16, lda = K0PAD
__global__ void build_edge_feat(const unsigned short* __restrict__ h2b,
                                const float* __restrict__ eattr, const int* __restrict__ src,
                                const int* __restrict__ dst, unsigned short* __restrict__ ef) {
  int e = blockIdx.x;
  int s = src[e], d = dst[e];
  for (int c = threadIdx.x; c < K0PAD; c += blockDim.x) {
    unsigned short v;
    if (c < HDIM)            v = h2b[(size_t)s * HDIM + c];
    else if (c < 2 * HDIM)   v = h2b[(size_t)d * HDIM + (c - HDIM)];
    else if (c < 2*HDIM + 3) v = f32_to_bf16_rte(eattr[(size_t)e * 3 + (c - 2*HDIM)]);
    else                     v = 0;
    ef[(size_t)e * K0PAD + c] = v;
  }
}

// ---------- sequential GRU scan: single 768-thread workgroup ----------
// gi: [T, 768] f32 precomputed input projections; Whh [768,256] f32 (L2-resident);
// h lives in LDS (broadcast reads), each thread owns one gh row (256 FMAs/step).
__global__ void gru_scan(const float* __restrict__ gi, const float* __restrict__ Whh,
                         const float* __restrict__ bhh, unsigned short* __restrict__ hsb,
                         int T) {
  __shared__ __align__(16) float h[GDIM];
  __shared__ float gh[TGATE];
  int t = threadIdx.x;                      // 0..767
  if (t < GDIM) h[t] = 0.f;
  float bh = bhh[t];
  const float4* wr = (const float4*)(Whh + (size_t)t * GDIM);
  __syncthreads();
  for (int step = 0; step < T; ++step) {
    const float4* h4 = (const float4*)h;
    float acc = 0.f;
    #pragma unroll 8
    for (int k = 0; k < GDIM / 4; ++k) {
      float4 w = wr[k];
      float4 hv = h4[k];
      acc = fmaf(w.x, hv.x, fmaf(w.y, hv.y, fmaf(w.z, hv.z, fmaf(w.w, hv.w, acc))));
    }
    gh[t] = acc + bh;
    __syncthreads();
    if (t < GDIM) {
      const float* git = gi + (size_t)step * TGATE;
      float r  = sigmoidf_(git[t] + gh[t]);
      float zt = sigmoidf_(git[GDIM + t] + gh[GDIM + t]);
      float nv = tanhf(git[2 * GDIM + t] + r * gh[2 * GDIM + t]);
      float hn = (1.f - zt) * nv + zt * h[t];
      h[t] = hn;
      hsb[(size_t)step * GDIM + t] = f32_to_bf16_rte(hn);
    }
    __syncthreads();
  }
}

// ---------- final 128-dot matvec ----------
__global__ void lin2_out(const float* __restrict__ L1, const float* __restrict__ W,
                         const float* __restrict__ b, float* __restrict__ out, int n) {
  int i = blockIdx.x * blockDim.x + threadIdx.x;
  if (i >= n) return;
  const float* r = L1 + (size_t)i * 128;
  float acc = b[0];
  #pragma unroll 16
  for (int j = 0; j < 128; ++j) acc = fmaf(r[j], W[j], acc);
  out[i] = acc;
}

// ---------- launcher ----------
extern "C" void kernel_launch(void* const* d_in, const int* in_sizes, int n_in,
                              void* d_out, int out_size, void* d_ws, size_t ws_size,
                              hipStream_t stream) {
  (void)in_sizes; (void)n_in; (void)out_size; (void)ws_size;
  const float* x       = (const float*)d_in[0];
  const int*   ei      = (const int*)d_in[1];
  const int*   src     = ei;
  const int*   dst     = ei + N_EDGES;
  const float* eattr   = (const float*)d_in[2];
  const float* W1      = (const float*)d_in[3];
  const float* att_s1  = (const float*)d_in[4];
  const float* att_d1  = (const float*)d_in[5];
  const float* b1      = (const float*)d_in[6];
  const float* gm1     = (const float*)d_in[7];
  const float* bt1     = (const float*)d_in[8];
  const float* mu1     = (const float*)d_in[9];
  const float* vr1     = (const float*)d_in[10];
  const float* W2      = (const float*)d_in[11];
  const float* att_s2  = (const float*)d_in[12];
  const float* att_d2  = (const float*)d_in[13];
  const float* b2      = (const float*)d_in[14];
  const float* gm2     = (const float*)d_in[15];
  const float* bt2     = (const float*)d_in[16];
  const float* mu2     = (const float*)d_in[17];
  const float* vr2     = (const float*)d_in[18];
  const float* Wih0    = (const float*)d_in[19];
  const float* Whh0    = (const float*)d_in[20];
  const float* bih0    = (const float*)d_in[21];
  const float* bhh0    = (const float*)d_in[22];
  const float* Wih1    = (const float*)d_in[23];
  const float* Whh1    = (const float*)d_in[24];
  const float* bih1    = (const float*)d_in[25];
  const float* bhh1    = (const float*)d_in[26];
  const float* Wlin1   = (const float*)d_in[27];
  const float* blin1   = (const float*)d_in[28];
  const float* Wlin2   = (const float*)d_in[29];
  const float* blin2   = (const float*)d_in[30];

  char* base = (char*)d_ws;
  size_t off = 0;
  auto take = [&](size_t bytes) -> void* {
    void* p = base + off;
    off += (bytes + 255) & ~(size_t)255;
    return p;
  };
  float*          cbuf  = (float*)take(2 * 4);
  float*          a_src = (float*)take((size_t)N_NODES * 4);
  float*          a_dst = (float*)take((size_t)N_NODES * 4);
  unsigned*       mmap  = (unsigned*)take((size_t)N_NODES * 4);
  float*          zbuf  = (float*)take((size_t)N_NODES * 4);
  float*          ebuf  = (float*)take((size_t)(N_EDGES + N_NODES) * 4);
  float*          agg   = (float*)take((size_t)N_NODES * 4);
  unsigned short* h1b   = (unsigned short*)take((size_t)N_NODES * HDIM * 2);
  unsigned short* W2b   = (unsigned short*)take((size_t)HDIM * HDIM * 2);
  float*          xw2   = (float*)take((size_t)N_NODES * HDIM * 4);
  float*          h2acc = (float*)take((size_t)N_NODES * HDIM * 4);
  unsigned short* h2b   = (unsigned short*)take((size_t)N_NODES * HDIM * 2);
  unsigned short* Wih0T = (unsigned short*)take((size_t)K0PAD * TGATE * 2);
  unsigned short* Wih1T = (unsigned short*)take((size_t)GDIM * TGATE * 2);
  unsigned short* Wl1T  = (unsigned short*)take((size_t)GDIM * 128 * 2);
  unsigned short* efb   = (unsigned short*)take((size_t)N_EDGES * K0PAD * 2);
  float*          gi    = (float*)take((size_t)N_EDGES * TGATE * 4);
  unsigned short* g0b   = (unsigned short*)take((size_t)N_EDGES * GDIM * 2);
  unsigned short* g1b   = (unsigned short*)take((size_t)N_EDGES * GDIM * 2);
  float*          l1f   = (float*)take((size_t)N_EDGES * 128 * 4);

  const int EN = N_EDGES + N_NODES;
  dim3 b256(256), gemm_blk(64);   // 2 waves per GEMM block

  // weight prep (bf16 K x N panels)
  conv_bf16<<<dim3((HDIM * HDIM + 255) / 256), b256, 0, stream>>>(W2, W2b, HDIM * HDIM);
  transpose_w<<<dim3((K0PAD * TGATE + 255) / 256), b256, 0, stream>>>(Wih0, Wih0T, TGATE, 2 * HDIM + 3, K0PAD);
  transpose_w<<<dim3((GDIM * TGATE + 255) / 256), b256, 0, stream>>>(Wih1, Wih1T, TGATE, GDIM, GDIM);
  transpose_w<<<dim3((GDIM * 128 + 255) / 256), b256, 0, stream>>>(Wlin1, Wl1T, 128, GDIM, GDIM);

  // ---- GAT layer 1 (rank-1) ----
  zero_u32<<<dim3((N_NODES + 255) / 256), b256, 0, stream>>>(mmap, N_NODES);
  zero_u32<<<dim3((N_NODES + 255) / 256), b256, 0, stream>>>((unsigned*)zbuf, N_NODES);
  zero_u32<<<dim3((N_NODES + 255) / 256), b256, 0, stream>>>((unsigned*)agg, N_NODES);
  dot2_256<<<dim3(1), b256, 0, stream>>>(W1, att_s1, att_d1, cbuf);
  scale_scores<<<dim3((N_NODES + 255) / 256), b256, 0, stream>>>(x, cbuf, a_src, a_dst, N_NODES);
  edge_lrelu_max<<<dim3((EN + 255) / 256), b256, 0, stream>>>(a_src, a_dst, src, dst, mmap, N_EDGES, N_NODES);
  edge_exp_sum<<<dim3((EN + 255) / 256), b256, 0, stream>>>(a_src, a_dst, src, dst, mmap, ebuf, zbuf, N_EDGES, N_NODES);
  gat1_aggregate<<<dim3((EN + 255) / 256), b256, 0, stream>>>(x, src, dst, ebuf, zbuf, agg, N_EDGES, N_NODES);
  gat1_finalize<<<dim3(N_NODES), b256, 0, stream>>>(agg, W1, b1, gm1, bt1, mu1, vr1, h1b);

  // ---- GAT layer 2 ----
  // xw2 = h1 @ W2 : WMMA bf16, M=20000, N=256, K=256 -> grid (256/128, 20000/32)
  wmma_gemm_bf16<<<dim3(HDIM / 128, N_NODES / 32), gemm_blk, 0, stream>>>(
      h1b, HDIM, W2b, HDIM, nullptr, xw2, HDIM, HDIM, 0);
  rowdot2<<<dim3(N_NODES), b256, 0, stream>>>(xw2, att_s2, att_d2, a_src, a_dst);
  zero_u32<<<dim3((N_NODES + 255) / 256), b256, 0, stream>>>(mmap, N_NODES);
  zero_u32<<<dim3((N_NODES + 255) / 256), b256, 0, stream>>>((unsigned*)zbuf, N_NODES);
  zero_u32<<<dim3(((unsigned)((size_t)N_NODES * HDIM) + 255) / 256), b256, 0, stream>>>(
      (unsigned*)h2acc, (unsigned)((size_t)N_NODES * HDIM));
  edge_lrelu_max<<<dim3((EN + 255) / 256), b256, 0, stream>>>(a_src, a_dst, src, dst, mmap, N_EDGES, N_NODES);
  edge_exp_sum<<<dim3((EN + 255) / 256), b256, 0, stream>>>(a_src, a_dst, src, dst, mmap, ebuf, zbuf, N_EDGES, N_NODES);
  gat2_scatter<<<dim3(EN), b256, 0, stream>>>(xw2, src, dst, ebuf, zbuf, h2acc, N_EDGES);
  gat2_finalize<<<dim3(N_NODES), b256, 0, stream>>>(h2acc, b2, gm2, bt2, mu2, vr2, h2b);

  // ---- edge features + GRU stack ----
  build_edge_feat<<<dim3(N_EDGES), b256, 0, stream>>>(h2b, eattr, src, dst, efb);
  // gi0 = ef @ Wih0^T + bih0 : M=65536, K=544, N=768 (~55 GFLOP, WMMA bf16)
  wmma_gemm_bf16<<<dim3(TGATE / 128, N_EDGES / 32), gemm_blk, 0, stream>>>(
      efb, K0PAD, Wih0T, TGATE, bih0, gi, TGATE, K0PAD, 0);
  gru_scan<<<dim3(1), dim3(TGATE), 0, stream>>>(gi, Whh0, bhh0, g0b, N_EDGES);
  // gi1 = g0 @ Wih1^T + bih1 : M=65536, K=256, N=768 (reuses gi buffer)
  wmma_gemm_bf16<<<dim3(TGATE / 128, N_EDGES / 32), gemm_blk, 0, stream>>>(
      g0b, GDIM, Wih1T, TGATE, bih1, gi, TGATE, GDIM, 0);
  gru_scan<<<dim3(1), dim3(TGATE), 0, stream>>>(gi, Whh1, bhh1, g1b, N_EDGES);

  // ---- head ----
  wmma_gemm_bf16<<<dim3(128 / 128, N_EDGES / 32), gemm_blk, 0, stream>>>(
      g1b, GDIM, Wl1T, 128, blin1, l1f, 128, GDIM, 1);
  lin2_out<<<dim3((N_EDGES + 255) / 256), b256, 0, stream>>>(l1f, Wlin2, blin2, (float*)d_out, N_EDGES);
}
